// MultiheadAttention_81741817577826
// MI455X (gfx1250) — matrix operand order repaired and checked by reading, not compiled, not run on hardware
//
#include <hip/hip_runtime.h>

// ---------------------------------------------------------------------------
// Types for CDNA5 WMMA (wave32): 16x16x32 bf16 -> f32
// ---------------------------------------------------------------------------
typedef __attribute__((ext_vector_type(16))) __bf16 v16bf;
typedef __attribute__((ext_vector_type(8)))  float  v8f;
typedef __attribute__((ext_vector_type(4)))  unsigned int u32x4;
typedef __attribute__((ext_vector_type(8)))  int i32x8;
typedef __attribute__((ext_vector_type(4)))  int i32x4;

union Frag16 { uint4 u[2]; v16bf v; };
union U4U16  { uint4 u; unsigned short s[8]; };

static __device__ __forceinline__ unsigned short f32_to_bf16(float f) {
  unsigned int u = __float_as_uint(f);
  u += 0x7FFFu + ((u >> 16) & 1u);   // round-to-nearest-even
  return (unsigned short)(u >> 16);
}

static __device__ __forceinline__ v8f wmma_bf16(v16bf a, v16bf b, v8f c) {
  return __builtin_amdgcn_wmma_f32_16x16x32_bf16(false, a, false, b, (short)0, c,
                                                 false, false);
}

// Low 32 bits of a generic pointer to LDS == LDS byte offset (ISA 10.2 aperture map)
static __device__ __forceinline__ unsigned lds_off_u32(const void* p) {
  return (unsigned)(unsigned long long)p;
}

// CDNA5 async global->LDS copy (ASYNCcnt-tracked). VDST = LDS offset, GV mode.
static __device__ __forceinline__ void async_load_lds_b128(const void* gaddr,
                                                           unsigned lds_off) {
  asm volatile("global_load_async_to_lds_b128 %0, %1, off"
               :: "v"(lds_off), "v"(gaddr)
               : "memory");
}
static __device__ __forceinline__ void wait_async0() {
  asm volatile("s_wait_asynccnt 0" ::: "memory");
}

// ---------------------------------------------------------------------------
// Tensor Data Mover: one-instruction 2D tile DMA global->LDS (TENSORcnt).
// D# per ISA 8.3/8.4: group0 = {flags, lds_addr, global_addr, type=2},
// group1 = {data_size=2B, tensor dims, tile dims, dim0 stride}. Tile is written
// to LDS tile-row-major, matching the WMMA fragment loaders below.
// This toolchain exposes the 6-arg builtin: (g0, g1, g2, g3, g4, cpol).
// ---------------------------------------------------------------------------
#if __has_builtin(__builtin_amdgcn_tensor_load_to_lds)
#define HAVE_TDM 1
static __device__ __forceinline__ void tdm_load_2d(
    unsigned lds_addr, const void* gaddr,
    unsigned tensor_d0, unsigned tensor_d1,   // tensor extent (elements)
    unsigned tile_d0, unsigned tile_d1,       // tile extent (elements)
    unsigned stride_d0) {                     // row stride (elements)
  const unsigned long long ga = (unsigned long long)gaddr;
  u32x4 g0;
  g0[0] = 1u;                                          // count=1, user-mode
  g0[1] = lds_addr;                                    // LDS byte address
  g0[2] = (unsigned)(ga & 0xFFFFFFFFull);              // global_addr[31:0]
  g0[3] = (unsigned)((ga >> 32) & 0x01FFFFFFull)       // global_addr[56:32]
        | (2u << 30);                                  // type = 2 ("image")
  i32x8 g1;
  g1[0] = (int)(1u << 16);                             // data_size=1 (2 bytes)
  g1[1] = (int)((tensor_d0 & 0xFFFFu) << 16);          // tensor_dim0[15:0]
  g1[2] = (int)((tensor_d0 >> 16) | ((tensor_d1 & 0xFFFFu) << 16));
  g1[3] = (int)((tensor_d1 >> 16) | (tile_d0 << 16));  // dim1 hi | tile_dim0
  g1[4] = (int)(tile_d1 & 0xFFFFu);                    // tile_dim1 (tile_dim2=0)
  g1[5] = (int)stride_d0;                              // tensor_dim0_stride[31:0]
  g1[6] = 0;
  g1[7] = 0;
  i32x4 z4 = {0, 0, 0, 0};
  i32x8 z8 = {0, 0, 0, 0, 0, 0, 0, 0};
  __builtin_amdgcn_tensor_load_to_lds(g0, g1, z4, z4, z8, 0);
}
static __device__ __forceinline__ void wait_tensor0() {
  __builtin_amdgcn_s_wait_tensorcnt(0);
}
#else
#define HAVE_TDM 0
#endif

// A-matrix 16x32 bf16 fragment (ISA 7.12.2): lane m = lane&15, hg = lane>>4;
// VGPR0-3 hold K = hg*8 + 0..7, VGPR4-7 hold K = 16 + hg*8 + 0..7.
static __device__ __forceinline__ v16bf load_frag_a(const unsigned short* base,
                                                    int row0, int stride, int lane) {
  const int m = lane & 15, hg = lane >> 4;
  const unsigned short* p = base + (size_t)(row0 + m) * stride;
  Frag16 f;
  f.u[0] = *(const uint4*)(p + hg * 8);
  f.u[1] = *(const uint4*)(p + 16 + hg * 8);
  return f.v;
}

// B-matrix 32x16 bf16 fragment from B^T-row-major storage base[n*stride+k]:
// lane n = lane&15; VGPR0-3 hold K = hg*8 + 0..7, VGPR4-7 hold K = 16 + hg*8 + 0..7.
static __device__ __forceinline__ v16bf load_frag_b(const unsigned short* base,
                                                    int col0, int stride, int lane) {
  const int n = lane & 15, hg = lane >> 4;
  const unsigned short* p = base + (size_t)(col0 + n) * stride;
  Frag16 f;
  f.u[0] = *(const uint4*)(p + hg * 8);
  f.u[1] = *(const uint4*)(p + 16 + hg * 8);
  return f.v;
}

// ---------------------------------------------------------------------------
// fp32 -> bf16 cast
// ---------------------------------------------------------------------------
__global__ void cast_f32_to_bf16(const float* __restrict__ src,
                                 unsigned short* __restrict__ dst, int n) {
  int i = blockIdx.x * blockDim.x + threadIdx.x;
  if (i < n) dst[i] = f32_to_bf16(src[i]);
}

// ---------------------------------------------------------------------------
// RoPE over full hidden dim (matches reference) fused with bf16 downcast
// ---------------------------------------------------------------------------
__global__ void rope_f32_to_bf16(const float* __restrict__ src,
                                 unsigned short* __restrict__ dst,
                                 int total_pairs, int S, int D) {
  int i = blockIdx.x * blockDim.x + threadIdx.x;
  if (i >= total_pairs) return;
  const int hp = D >> 1;
  int row = i / hp;
  int j = i - row * hp;
  int s = row & (S - 1);
  float inv = __powf(10000.0f, (-2.0f * (float)j) / (float)D);
  float ang = (float)s * inv;
  float sn, cs;
  __sincosf(ang, &sn, &cs);
  float x0 = src[2 * i], x1 = src[2 * i + 1];
  dst[2 * i]     = f32_to_bf16(x0 * cs - x1 * sn);
  dst[2 * i + 1] = f32_to_bf16(x0 * sn + x1 * cs);
}

// ---------------------------------------------------------------------------
// Tiled bf16 WMMA GEMM: C[M,N] = A[M,K] * W[N,K]^T
// 128x128 block tile, K-step 32, double-buffered TDM tile staging (wave 0
// issues one tensor_load_to_lds per tile; TENSORcnt drain + barrier publishes).
// 8 waves arranged 2(M)x4(N), each wave 4x2 WMMA tiles.
// mode 0: f32 output; mode 1: bf16 output.
// ---------------------------------------------------------------------------
#define GTM 128
#define GTN 128
#define GTK 32

__global__ __launch_bounds__(256) void gemm_bf16_wmma(
    const unsigned short* __restrict__ A,
    const unsigned short* __restrict__ W,
    float* __restrict__ Cf,
    unsigned short* __restrict__ Cb,
    int M, int N, int K, int mode) {
  __shared__ unsigned short As[2][GTM * GTK];
  __shared__ unsigned short Ws[2][GTN * GTK];

  const int tid  = threadIdx.x;
  const int lane = tid & 31;
  const int w    = tid >> 5;
  const int wm   = w >> 2;        // 0..1 -> 64 rows
  const int wn   = w & 3;         // 0..3 -> 32 cols
  const int bm   = blockIdx.x * GTM;
  const int bn   = blockIdx.y * GTN;

#if HAVE_TDM
  // stage one K-tile into buffer `buf` via TDM (wave 0 only)
  auto stage = [&](int buf, int k0) {
    if (w == 0) {
      tdm_load_2d(lds_off_u32(&As[buf][0]), A + (size_t)bm * K + k0,
                  (unsigned)(K - k0), (unsigned)GTM, GTK, GTM, (unsigned)K);
      tdm_load_2d(lds_off_u32(&Ws[buf][0]), W + (size_t)bn * K + k0,
                  (unsigned)(K - k0), (unsigned)GTN, GTK, GTN, (unsigned)K);
    }
  };
  auto drain = [&]() { if (w == 0) wait_tensor0(); };
#else
  // fallback: per-thread async global->LDS copies (ASYNCcnt)
  auto stage = [&](int buf, int k0) {
#pragma unroll
    for (int i = 0; i < 2; ++i) {
      int idx = tid + i * 256;
      int r = idx >> 2;
      int c = (idx & 3) * 8;
      async_load_lds_b128(A + (size_t)(bm + r) * K + k0 + c,
                          lds_off_u32(&As[buf][r * GTK + c]));
      async_load_lds_b128(W + (size_t)(bn + r) * K + k0 + c,
                          lds_off_u32(&Ws[buf][r * GTK + c]));
    }
  };
  auto drain = [&]() { wait_async0(); };
#endif

  v8f acc[4][2];
#pragma unroll
  for (int i = 0; i < 4; ++i)
#pragma unroll
    for (int j = 0; j < 2; ++j)
#pragma unroll
      for (int r = 0; r < 8; ++r) acc[i][j][r] = 0.f;

  stage(0, 0);
  drain();
  __syncthreads();

  const int nk = K / GTK;
  for (int t = 0; t < nk; ++t) {
    const int cur = t & 1;
    if (t + 1 < nk) stage(cur ^ 1, (t + 1) * GTK);   // DMA next tile during math

    v16bf bf0 = load_frag_b(&Ws[cur][0], wn * 32,      GTK, lane);
    v16bf bf1 = load_frag_b(&Ws[cur][0], wn * 32 + 16, GTK, lane);
#pragma unroll
    for (int i = 0; i < 4; ++i) {
      v16bf af = load_frag_a(&As[cur][0], wm * 64 + i * 16, GTK, lane);
      acc[i][0] = wmma_bf16(af, bf0, acc[i][0]);
      acc[i][1] = wmma_bf16(af, bf1, acc[i][1]);
    }

    if (t + 1 < nk) drain();
    __syncthreads();
  }

  // C/D layout: lane -> N = lane&15; VGPR r -> M = r + 8*(lane>>4)
  const int cn = lane & 15, rhi = (lane >> 4) * 8;
#pragma unroll
  for (int i = 0; i < 4; ++i)
#pragma unroll
    for (int j = 0; j < 2; ++j)
#pragma unroll
      for (int r = 0; r < 8; ++r) {
        int m = bm + wm * 64 + i * 16 + rhi + r;
        int n = bn + wn * 32 + j * 16 + cn;
        float vv = acc[i][j][r];
        if (mode == 0) Cf[(size_t)m * N + n] = vv;
        else           Cb[(size_t)m * N + n] = f32_to_bf16(vv);
      }
}

// ---------------------------------------------------------------------------
// Flash attention (online softmax), all matmuls via WMMA.
// Block = one (b, h, 64-row Q tile). 32-row K/V chunks, double-buffered:
// K chunk staged via async global->LDS; V chunk prefetched to VGPRs and
// transposed into LDS at iteration end.
// ---------------------------------------------------------------------------
#define ATT_BQ 64
#define ATT_BK 32
#define ATT_HD 128
#define ATT_S  1024
#define ATT_D  2048
#define ATT_NC (ATT_S / ATT_BK)

__global__ __launch_bounds__(256) void attn_flash_wmma(
    const unsigned short* __restrict__ Q,
    const unsigned short* __restrict__ Kb,
    const unsigned short* __restrict__ Vb,
    unsigned short* __restrict__ Ctx) {
  __shared__ unsigned short qs [ATT_BQ * ATT_HD];      // 16 KB
  __shared__ unsigned short ks [2][ATT_BK * ATT_HD];   // 16 KB (rows = key idx)
  __shared__ unsigned short vsT[2][ATT_HD * ATT_BK];   // 16 KB (rows = head dim)
  __shared__ float          ss [ATT_BQ * ATT_BK];      //  8 KB scores f32
  __shared__ unsigned short ps [ATT_BQ * ATT_BK];      //  4 KB probabilities bf16
  __shared__ float red[ATT_BQ * 4];                    //  1 KB partial reductions
  __shared__ float rowm[ATT_BQ], rowl[ATT_BQ], rowsc[ATT_BQ];

  const int tid = threadIdx.x, lane = tid & 31, w = tid >> 5;
  const int qt = blockIdx.x;            // 0..15
  const int bh = blockIdx.y;            // 0..127
  const int b = bh >> 4, h = bh & 15;

  const size_t qbase = ((size_t)b * ATT_S + qt * ATT_BQ) * ATT_D + h * ATT_HD;
  const size_t bhbase = (size_t)b * ATT_S * ATT_D + h * ATT_HD;

  // per-thread staging coordinates for 32x128 chunks (2 x b128 per thread)
  const int sr0 = tid >> 4,               sc0 = (tid & 15) * 8;
  const int sr1 = (tid + 256) >> 4,       sc1 = ((tid + 256) & 15) * 8;

  auto stage_k = [&](int buf, int kc) {
    const size_t kb0 = bhbase + (size_t)(kc * ATT_BK) * ATT_D;
    async_load_lds_b128(Kb + kb0 + (size_t)sr0 * ATT_D + sc0,
                        lds_off_u32(&ks[buf][sr0 * ATT_HD + sc0]));
    async_load_lds_b128(Kb + kb0 + (size_t)sr1 * ATT_D + sc1,
                        lds_off_u32(&ks[buf][sr1 * ATT_HD + sc1]));
  };
  auto load_v = [&](int kc, uint4* vr) {
    const size_t vb0 = bhbase + (size_t)(kc * ATT_BK) * ATT_D;
    vr[0] = *(const uint4*)(Vb + vb0 + (size_t)sr0 * ATT_D + sc0);
    vr[1] = *(const uint4*)(Vb + vb0 + (size_t)sr1 * ATT_D + sc1);
  };
  auto store_vT = [&](int buf, const uint4* vr) {
    U4U16 u0; u0.u = vr[0];
    U4U16 u1; u1.u = vr[1];
#pragma unroll
    for (int e = 0; e < 8; ++e) {
      vsT[buf][(sc0 + e) * ATT_BK + sr0] = u0.s[e];
      vsT[buf][(sc1 + e) * ATT_BK + sr1] = u1.s[e];
    }
  };

  // stage Q tile (64x128) asynchronously
#pragma unroll
  for (int i = 0; i < 4; ++i) {
    int idx = tid + i * 256;
    int r = idx >> 4;                   // 0..63
    int c = (idx & 15) * 8;             // 0..120
    async_load_lds_b128(Q + qbase + (size_t)r * ATT_D + c,
                        lds_off_u32(&qs[r * ATT_HD + c]));
  }
  if (tid < ATT_BQ) { rowm[tid] = -3.0e38f; rowl[tid] = 0.f; }

  v8f cacc[4];
#pragma unroll
  for (int t = 0; t < 4; ++t)
#pragma unroll
    for (int r = 0; r < 8; ++r) cacc[t][r] = 0.f;

  const int wm = w >> 1;                // 0..3 : 16-row slice of Q tile
  const int wn = w & 1;                 // 0..1 : S cols (16) / ctx cols (64)
  const int cn = lane & 15, rhi = (lane >> 4) * 8;
  const int rrow = tid >> 2, rq = tid & 3;  // softmax: 4 threads per row
  const float scale = 0.02209708691f;       // 1/sqrt(2048)

  // prologue: chunk 0 K async + V regs, then commit V transpose
  uint4 vreg[2], vregN[2];
  stage_k(0, 0);
  load_v(0, vreg);
  store_vT(0, vreg);
  wait_async0();
  __syncthreads();

  for (int kc = 0; kc < ATT_NC; ++kc) {
    const int cur = kc & 1;
    const bool more = (kc + 1 < ATT_NC);
    if (more) {
      stage_k(cur ^ 1, kc + 1);   // async DMA next K chunk
      load_v(kc + 1, vregN);      // prefetch next V chunk to VGPRs
    }

    // S = Q * Kc^T  (64x32), wave tile (wm, wn)
    {
      v8f s;
#pragma unroll
      for (int r = 0; r < 8; ++r) s[r] = 0.f;
#pragma unroll
      for (int d0 = 0; d0 < ATT_HD; d0 += 32) {
        v16bf af = load_frag_a(qs + d0, wm * 16, ATT_HD, lane);
        v16bf bf = load_frag_b(&ks[cur][0] + d0, wn * 16, ATT_HD, lane);
        s = wmma_bf16(af, bf, s);
      }
#pragma unroll
      for (int r = 0; r < 8; ++r)
        ss[(wm * 16 + rhi + r) * ATT_BK + wn * 16 + cn] = s[r] * scale;
    }
    __syncthreads();

    // online softmax: 4 threads per row (partial max -> combine -> exp/sum)
    {
      float pm = -3.0e38f;
#pragma unroll
      for (int j = 0; j < 8; ++j) pm = fmaxf(pm, ss[rrow * ATT_BK + rq * 8 + j]);
      red[rrow * 4 + rq] = pm;
      __syncthreads();
      if (tid < ATT_BQ) {
        float mx = fmaxf(fmaxf(red[tid * 4], red[tid * 4 + 1]),
                         fmaxf(red[tid * 4 + 2], red[tid * 4 + 3]));
        float m0 = rowm[tid];
        mx = fmaxf(m0, mx);
        rowsc[tid] = __expf(m0 - mx);
        rowm[tid] = mx;
      }
      __syncthreads();
      float mx = rowm[rrow];
      float psum = 0.f;
#pragma unroll
      for (int j = 0; j < 8; ++j) {
        float pv = __expf(ss[rrow * ATT_BK + rq * 8 + j] - mx);
        ps[rrow * ATT_BK + rq * 8 + j] = f32_to_bf16(pv);
        psum += pv;
      }
      red[rrow * 4 + rq] = psum;
      __syncthreads();
      if (tid < ATT_BQ)
        rowl[tid] = rowl[tid] * rowsc[tid] +
                    red[tid * 4] + red[tid * 4 + 1] +
                    red[tid * 4 + 2] + red[tid * 4 + 3];
      __syncthreads();
    }

    // rescale running ctx and accumulate P * Vc (ctx tile: rows 16*wm, cols 64*wn)
    {
      float scl[8];
#pragma unroll
      for (int r = 0; r < 8; ++r) scl[r] = rowsc[wm * 16 + rhi + r];
      v16bf af = load_frag_a(ps, wm * 16, ATT_BK, lane);
#pragma unroll
      for (int t = 0; t < 4; ++t) {
#pragma unroll
        for (int r = 0; r < 8; ++r) cacc[t][r] *= scl[r];
        v16bf bf = load_frag_b(&vsT[cur][0], wn * 64 + t * 16, ATT_BK, lane);
        cacc[t] = wmma_bf16(af, bf, cacc[t]);
      }
    }

    if (more) {
      store_vT(cur ^ 1, vregN);   // commit next V chunk transpose
      wait_async0();              // drain next K chunk DMA
    }
    __syncthreads();
  }

  // finalize: ctx /= l, write bf16 back in [b, s, h*hd + n] layout
#pragma unroll
  for (int t = 0; t < 4; ++t)
#pragma unroll
    for (int r = 0; r < 8; ++r) {
      int m = wm * 16 + rhi + r;
      int n = wn * 64 + t * 16 + cn;
      float vv = cacc[t][r] / rowl[m];
      Ctx[qbase + (size_t)m * ATT_D + n] = f32_to_bf16(vv);
    }
}

// ---------------------------------------------------------------------------
// Orchestration
// ---------------------------------------------------------------------------
extern "C" void kernel_launch(void* const* d_in, const int* in_sizes, int n_in,
                              void* d_out, int out_size, void* d_ws, size_t ws_size,
                              hipStream_t stream) {
  const float* x  = (const float*)d_in[0];
  const float* wq = (const float*)d_in[1];
  const float* wk = (const float*)d_in[2];
  const float* wv = (const float*)d_in[3];
  const float* wo = (const float*)d_in[4];

  const int B = 8, S = 1024, D = 2048;
  const int M = B * S;                       // 8192
  const size_t nX = (size_t)M * D;           // 16,777,216
  const size_t nW = (size_t)D * D;           //  4,194,304

  char* p = (char*)d_ws;
  unsigned short* xb  = (unsigned short*)p; p += nX * 2;   // 32 MB (reused as ctx)
  unsigned short* wqb = (unsigned short*)p; p += nW * 2;   //  8 MB
  unsigned short* wkb = (unsigned short*)p; p += nW * 2;
  unsigned short* wvb = (unsigned short*)p; p += nW * 2;
  unsigned short* wob = (unsigned short*)p; p += nW * 2;
  float*          qf  = (float*)p;          p += nX * 4;   // 64 MB f32 scratch
  unsigned short* qb  = (unsigned short*)p; p += nX * 2;   // 32 MB
  unsigned short* kb  = (unsigned short*)p; p += nX * 2;   // 32 MB
  unsigned short* vb  = (unsigned short*)p; p += nX * 2;   // 32 MB
  unsigned short* ctx = xb;                                // alias after projections

  dim3 blk(256);

  cast_f32_to_bf16<<<dim3((unsigned)((nX + 255) / 256)), blk, 0, stream>>>(x,  xb,  (int)nX);
  cast_f32_to_bf16<<<dim3((unsigned)((nW + 255) / 256)), blk, 0, stream>>>(wq, wqb, (int)nW);
  cast_f32_to_bf16<<<dim3((unsigned)((nW + 255) / 256)), blk, 0, stream>>>(wk, wkb, (int)nW);
  cast_f32_to_bf16<<<dim3((unsigned)((nW + 255) / 256)), blk, 0, stream>>>(wv, wvb, (int)nW);
  cast_f32_to_bf16<<<dim3((unsigned)((nW + 255) / 256)), blk, 0, stream>>>(wo, wob, (int)nW);

  dim3 ggrid(M / GTM, D / GTN);
  const int pairs = (int)(nX / 2);
  dim3 rgrid((unsigned)((pairs + 255) / 256));

  // Q = x wq^T ; RoPE ; bf16
  gemm_bf16_wmma<<<ggrid, blk, 0, stream>>>(xb, wqb, qf, nullptr, M, D, D, 0);
  rope_f32_to_bf16<<<rgrid, blk, 0, stream>>>(qf, qb, pairs, S, D);
  // K = x wk^T ; RoPE ; bf16
  gemm_bf16_wmma<<<ggrid, blk, 0, stream>>>(xb, wkb, qf, nullptr, M, D, D, 0);
  rope_f32_to_bf16<<<rgrid, blk, 0, stream>>>(qf, kb, pairs, S, D);
  // V = x wv^T ; bf16 directly
  gemm_bf16_wmma<<<ggrid, blk, 0, stream>>>(xb, wvb, nullptr, vb, M, D, D, 1);

  // flash attention per (b, h, 64-row q tile)
  dim3 agrid(S / ATT_BQ, B * 16);
  attn_flash_wmma<<<agrid, blk, 0, stream>>>(qb, kb, vb, ctx);

  // out = ctx wo^T  (f32 to d_out)
  gemm_bf16_wmma<<<ggrid, blk, 0, stream>>>(ctx, wob, (float*)d_out, nullptr, M, D, D, 0);
}
